// MultiHeadAttention_18915035971698
// MI455X (gfx1250) — matrix operand order repaired and checked by reading, not compiled, not run on hardware
//
#include <hip/hip_runtime.h>
#include <hip/hip_bf16.h>

// ---------------------------------------------------------------------------
// MHA forward for MI455X (gfx1250, wave32, WMMA, async-to-LDS staging).
// Pipeline: f32->bf16 converts -> QKV GEMMs (bf16 WMMA, double-buffered async
// LDS staging) -> RoPE -> flash attention (bf16 WMMA, online softmax, async
// K staging) -> output GEMM (f32 out).
// ---------------------------------------------------------------------------

typedef __bf16 bf16;
typedef bf16  bf16x16 __attribute__((ext_vector_type(16)));
typedef float f32x8   __attribute__((ext_vector_type(8)));

union Frag32B  { bf16x16 v; uint4 q[2]; };
union Chunk16B { uint4 q; bf16 e[8]; };

#define WMMA_BF16(A, B, C) \
  __builtin_amdgcn_wmma_f32_16x16x32_bf16(false, (A), false, (B), (short)0, (C), false, false)

// ---- CDNA5 async global->LDS staging (ASYNCcnt), with safe fallback -------
// Probe result: builtin exists with 4 args; arg0 is a non-const int4* in
// address space 1 (printed as "__device__" by clang in HIP mode).
#if defined(__gfx1250__) && __has_builtin(__builtin_amdgcn_global_load_async_to_lds_b128)
#define HAS_ASYNC_LDS 1
typedef int v4i __attribute__((vector_size(16)));
typedef __attribute__((address_space(1))) v4i gv4i;
typedef __attribute__((address_space(3))) v4i lv4i;
#define ASYNC_LOAD_B128(g, l) \
  __builtin_amdgcn_global_load_async_to_lds_b128((gv4i*)(g), (lv4i*)(l), 0, 0)
#else
#define HAS_ASYNC_LDS 0
#define ASYNC_LOAD_B128(g, l) (*(uint4*)(l) = *(const uint4*)(g))
#endif

template <int N>
__device__ __forceinline__ void wait_async_lds() {
#if defined(__gfx1250__) && __has_builtin(__builtin_amdgcn_s_wait_asynccnt)
  __builtin_amdgcn_s_wait_asynccnt(N);  // immediate operand required
#endif
}

__device__ __forceinline__ f32x8 zero8() {
  f32x8 z;
#pragma unroll
  for (int i = 0; i < 8; ++i) z[i] = 0.0f;
  return z;
}

// ---------------------------------------------------------------------------
// Elementwise f32 -> bf16 (vectorized by 4)
// ---------------------------------------------------------------------------
__global__ __launch_bounds__(256) void f32_to_bf16_kernel(const float* __restrict__ in,
                                                          bf16* __restrict__ out, int n) {
  int i = (blockIdx.x * 256 + threadIdx.x) * 4;
  if (i + 3 < n) {
    float4 f = *(const float4*)(in + i);
    out[i + 0] = (bf16)f.x;
    out[i + 1] = (bf16)f.y;
    out[i + 2] = (bf16)f.z;
    out[i + 3] = (bf16)f.w;
  }
}

// ---------------------------------------------------------------------------
// Generic bf16 GEMM: C[M,N] = A[M,K] * B[N,K]^T   (row-major, dims %128==0)
// 128x128 block tile, 256 threads = 8 waves (4x2), each wave 32x64 via 2x4
// WMMA tiles. K staged by 32 into double-buffered LDS via async-to-LDS.
// ---------------------------------------------------------------------------
template <typename OutT>
__global__ __launch_bounds__(256) void gemm_nt_bf16(const bf16* __restrict__ A,
                                                    const bf16* __restrict__ B,
                                                    OutT* __restrict__ C,
                                                    int M, int N, int K) {
  constexpr int AS = 40;  // LDS row stride (32 + 8 pad) in elements
  __shared__ bf16 sA[2][128 * AS];
  __shared__ bf16 sB[2][128 * AS];

  const int tid  = threadIdx.x;
  const int wave = tid >> 5;
  const int lane = tid & 31;
  const int l16  = lane & 15;
  const int hi   = lane >> 4;
  const int wr   = wave >> 1;  // 0..3 -> M offset wr*32
  const int wc   = wave & 1;   // 0..1 -> N offset wc*64
  const int blockM = blockIdx.y * 128;
  const int blockN = blockIdx.x * 128;

  f32x8 acc[2][4];
#pragma unroll
  for (int i = 0; i < 2; ++i)
#pragma unroll
    for (int j = 0; j < 4; ++j) acc[i][j] = zero8();

  const int r     = tid >> 1;        // 0..127: staged tile row
  const int halfo = (tid & 1) * 16;  // 0/16 elements: 32B half-row
  const bf16* aRow = A + (size_t)(blockM + r) * K + halfo;
  const bf16* bRow = B + (size_t)(blockN + r) * K + halfo;
  const int sOff = r * AS + halfo;

  // Prologue: stage tile 0 into buffer 0.
  ASYNC_LOAD_B128(aRow + 0, &sA[0][sOff]);
  ASYNC_LOAD_B128(aRow + 8, &sA[0][sOff + 8]);
  ASYNC_LOAD_B128(bRow + 0, &sB[0][sOff]);
  ASYNC_LOAD_B128(bRow + 8, &sB[0][sOff + 8]);

  const int nk = K >> 5;
  for (int i = 0; i < nk; ++i) {
    const int cur = i & 1;
    if (i + 1 < nk) {  // stage next tile into the other buffer (overlaps WMMA)
      const int nxt = cur ^ 1;
      const bf16* an = aRow + (size_t)(i + 1) * 32;
      const bf16* bn = bRow + (size_t)(i + 1) * 32;
      ASYNC_LOAD_B128(an + 0, &sA[nxt][sOff]);
      ASYNC_LOAD_B128(an + 8, &sA[nxt][sOff + 8]);
      ASYNC_LOAD_B128(bn + 0, &sB[nxt][sOff]);
      ASYNC_LOAD_B128(bn + 8, &sB[nxt][sOff + 8]);
      wait_async_lds<4>();  // in-order completion: current tile has landed
    } else {
      wait_async_lds<0>();
    }
    __syncthreads();

    const bf16* cA = sA[cur];
    const bf16* cB = sB[cur];
    // A frags: 16x32 bf16, lane(0..15)=M row, q[0]=K[kb..+7], q[1]=K[kb+16..+23]
    Frag32B af[2];
#pragma unroll
    for (int mf = 0; mf < 2; ++mf) {
      int mrow = wr * 32 + mf * 16 + l16;
      int kb   = hi * 8;
      af[mf].q[0] = *(const uint4*)&cA[mrow * AS + kb];
      af[mf].q[1] = *(const uint4*)&cA[mrow * AS + kb + 16];
    }
    // B frags: 32x16 bf16, lane(0..15)=N col, q[0]=K[kb..+7], q[1]=K[kb+16..+23]
    Frag32B bfr[4];
#pragma unroll
    for (int nf = 0; nf < 4; ++nf) {
      int nrow = wc * 64 + nf * 16 + l16;
      int kb   = hi * 8;
      bfr[nf].q[0] = *(const uint4*)&cB[nrow * AS + kb];
      bfr[nf].q[1] = *(const uint4*)&cB[nrow * AS + kb + 16];
    }
#pragma unroll
    for (int mf = 0; mf < 2; ++mf)
#pragma unroll
      for (int nf = 0; nf < 4; ++nf)
        acc[mf][nf] = WMMA_BF16(af[mf].v, bfr[nf].v, acc[mf][nf]);

    __syncthreads();  // all waves done reading buffer `cur` before reuse
  }

  // C layout: lane holds (M = v + 8*hi, N = l16) of each 16x16 tile
#pragma unroll
  for (int mf = 0; mf < 2; ++mf)
#pragma unroll
    for (int nf = 0; nf < 4; ++nf)
#pragma unroll
      for (int v = 0; v < 8; ++v) {
        int m = blockM + wr * 32 + mf * 16 + v + 8 * hi;
        int n = blockN + wc * 64 + nf * 16 + l16;
        C[(size_t)m * N + n] = (OutT)acc[mf][nf][v];
      }
}

// ---------------------------------------------------------------------------
// In-place RoPE on a bf16 [B*S, H*Dk] buffer. One thread per (even,odd) pair.
// ---------------------------------------------------------------------------
__global__ __launch_bounds__(256) void rope_bf16_kernel(bf16* __restrict__ t,
                                                        const float* __restrict__ pe,
                                                        int total_pairs) {
  int i = blockIdx.x * 256 + threadIdx.x;
  if (i >= total_pairs) return;
  int row = i >> 10;        // 1024 pairs per 2048-wide row
  int col = (i & 1023) * 2;
  int s   = row & 2047;     // row = b*2048 + s
  int d   = col & 127;      // position within head
  float ang = pe[s * 128 + d];
  float c = cosf(ang), sn = sinf(ang);
  size_t off = (size_t)row * 2048 + col;
  float u1 = (float)t[off];
  float u2 = (float)t[off + 1];
  t[off]     = (bf16)(u1 * c - u2 * sn);
  t[off + 1] = (bf16)(u2 * c + u1 * sn);
}

// ---------------------------------------------------------------------------
// Flash attention (causal). Grid: (S/128, B*H). 256 threads = 8 waves; wave w
// owns q-rows [w*16, w*16+16). Dynamic LDS 136KB: Q,K row-major (async
// staged); V transposed at store time; P tile for S->A-fragment re-layout.
// ---------------------------------------------------------------------------
__global__ __launch_bounds__(256) void flash_attn_kernel(const bf16* __restrict__ Q,
                                                         const bf16* __restrict__ K,
                                                         const bf16* __restrict__ V,
                                                         bf16* __restrict__ O,
                                                         int S) {
  extern __shared__ char smem[];
  constexpr int ST = 136;  // 128 + 8 pad
  bf16* sQ  = (bf16*)smem;
  bf16* sK  = sQ + 128 * ST;
  bf16* sVt = sK + 128 * ST;
  bf16* sP  = sVt + 128 * ST;

  const int tid  = threadIdx.x;
  const int wave = tid >> 5;
  const int lane = tid & 31;
  const int l16  = lane & 15;
  const int hi   = lane >> 4;
  const int qblk = blockIdx.x;
  const int bh   = blockIdx.y;
  const int b    = bh >> 4;
  const int h    = bh & 15;
  const int qbase = qblk * 128;
  const size_t ld = 2048;

  const bf16* Qb = Q + (size_t)b * S * ld + h * 128;
  const bf16* Kb = K + (size_t)b * S * ld + h * 128;
  const bf16* Vb = V + (size_t)b * S * ld + h * 128;

  // Stage Q tile (128x128 bf16) via async-to-LDS; completion covered by the
  // first loop iteration's wait_async + barrier.
#pragma unroll
  for (int i = 0; i < 8; ++i) {
    int c = tid + i * 256;
    int r = c >> 4, cc = c & 15;
    ASYNC_LOAD_B128(Qb + (size_t)(qbase + r) * ld + cc * 8, &sQ[r * ST + cc * 8]);
  }

  f32x8 oacc[8];
#pragma unroll
  for (int i = 0; i < 8; ++i) oacc[i] = zero8();
  float rowm[8], rowl[8];
#pragma unroll
  for (int v = 0; v < 8; ++v) { rowm[v] = -3.0e38f; rowl[v] = 0.0f; }

  const float scale = 0.08838834764831845f;  // 1/sqrt(128)

  for (int j = 0; j <= qblk; ++j) {
    const int kbase = j * 128;
    __syncthreads();  // all waves done with previous sK/sVt/sP
#pragma unroll
    for (int i = 0; i < 8; ++i) {
      int c = tid + i * 256;
      int r = c >> 4, cc = c & 15;
      // K tile: async global->LDS (row-major)
      ASYNC_LOAD_B128(Kb + (size_t)(kbase + r) * ld + cc * 8, &sK[r * ST + cc * 8]);
      // V tile: VGPR path with transpose into d-major sVt
      Chunk16B uv;
      uv.q = *(const uint4*)(Vb + (size_t)(kbase + r) * ld + cc * 8);
#pragma unroll
      for (int e = 0; e < 8; ++e) sVt[(cc * 8 + e) * ST + r] = uv.e[e];
    }
    wait_async_lds<0>();
    __syncthreads();

    // ---- S = Q * K^T (16 x 128 per wave; K-dim = 128) ----
    Frag32B aq[4];
#pragma unroll
    for (int ks = 0; ks < 4; ++ks) {
      int mrow = wave * 16 + l16;
      int kb   = ks * 32 + hi * 8;
      aq[ks].q[0] = *(const uint4*)&sQ[mrow * ST + kb];
      aq[ks].q[1] = *(const uint4*)&sQ[mrow * ST + kb + 16];
    }
    f32x8 sacc[8];
#pragma unroll
    for (int nt = 0; nt < 8; ++nt) {
      Frag32B bk[4];
#pragma unroll
      for (int ks = 0; ks < 4; ++ks) {  // batch loads ahead of the WMMA chain
        int nrow = nt * 16 + l16;
        int kb   = ks * 32 + hi * 8;
        bk[ks].q[0] = *(const uint4*)&sK[nrow * ST + kb];
        bk[ks].q[1] = *(const uint4*)&sK[nrow * ST + kb + 16];
      }
      sacc[nt] = zero8();
#pragma unroll
      for (int ks = 0; ks < 4; ++ks)
        sacc[nt] = WMMA_BF16(aq[ks].v, bk[ks].v, sacc[nt]);
    }

    // ---- scale + causal mask + online softmax ----
#pragma unroll
    for (int v = 0; v < 8; ++v) {
      int qi = qbase + wave * 16 + v + 8 * hi;
      float rmax = -3.0e38f;
#pragma unroll
      for (int nt = 0; nt < 8; ++nt) {
        int kj = kbase + nt * 16 + l16;
        float sv = sacc[nt][v] * scale;
        sv = (kj <= qi) ? sv : -3.0e38f;
        sacc[nt][v] = sv;
        rmax = fmaxf(rmax, sv);
      }
      rmax = fmaxf(rmax, __shfl_xor(rmax, 1));
      rmax = fmaxf(rmax, __shfl_xor(rmax, 2));
      rmax = fmaxf(rmax, __shfl_xor(rmax, 4));
      rmax = fmaxf(rmax, __shfl_xor(rmax, 8));
      float mnew = fmaxf(rowm[v], rmax);
      float corr = __expf(rowm[v] - mnew);
      rowm[v] = mnew;
      float rsum = 0.0f;
#pragma unroll
      for (int nt = 0; nt < 8; ++nt) {
        float p = __expf(sacc[nt][v] - mnew);
        sacc[nt][v] = p;
        rsum += p;
      }
      rsum += __shfl_xor(rsum, 1);
      rsum += __shfl_xor(rsum, 2);
      rsum += __shfl_xor(rsum, 4);
      rsum += __shfl_xor(rsum, 8);
      rowl[v] = rowl[v] * corr + rsum;
#pragma unroll
      for (int dt = 0; dt < 8; ++dt) oacc[dt][v] *= corr;
    }

    // ---- P (bf16) to LDS; each wave writes only its own 16 rows ----
#pragma unroll
    for (int nt = 0; nt < 8; ++nt)
#pragma unroll
      for (int v = 0; v < 8; ++v)
        sP[(wave * 16 + v + 8 * hi) * ST + nt * 16 + l16] = (bf16)sacc[nt][v];

    __syncthreads();  // P visible across lanes; all waves done reading sK

    // ---- O += P * V  (K-dim = 128 key positions) ----
    Frag32B ap2[4];
#pragma unroll
    for (int ks = 0; ks < 4; ++ks) {
      int mrow = wave * 16 + l16;
      int kb   = ks * 32 + hi * 8;
      ap2[ks].q[0] = *(const uint4*)&sP[mrow * ST + kb];
      ap2[ks].q[1] = *(const uint4*)&sP[mrow * ST + kb + 16];
    }
#pragma unroll
    for (int dt = 0; dt < 8; ++dt) {
      Frag32B bv[4];
#pragma unroll
      for (int ks = 0; ks < 4; ++ks) {  // batch loads ahead of the WMMA chain
        int drow = dt * 16 + l16;
        int kb   = ks * 32 + hi * 8;
        bv[ks].q[0] = *(const uint4*)&sVt[drow * ST + kb];
        bv[ks].q[1] = *(const uint4*)&sVt[drow * ST + kb + 16];
      }
#pragma unroll
      for (int ks = 0; ks < 4; ++ks)
        oacc[dt] = WMMA_BF16(ap2[ks].v, bv[ks].v, oacc[dt]);
    }
  }

  // ---- normalize and write O (bf16, [B*S, H*Dk]) ----
#pragma unroll
  for (int v = 0; v < 8; ++v) {
    float inv = 1.0f / rowl[v];
    int m = qbase + wave * 16 + v + 8 * hi;
    bf16* orow = O + (size_t)(b * S + m) * ld + h * 128;
#pragma unroll
    for (int dt = 0; dt < 8; ++dt) orow[dt * 16 + l16] = (bf16)(oacc[dt][v] * inv);
  }
}

// ---------------------------------------------------------------------------
// Host launcher
// ---------------------------------------------------------------------------
extern "C" void kernel_launch(void* const* d_in, const int* in_sizes, int n_in,
                              void* d_out, int out_size, void* d_ws, size_t ws_size,
                              hipStream_t stream) {
  constexpr int B = 2, S = 2048, E = 2048, H = 16, DK = 128;
  constexpr int M = B * S;    // 4096
  constexpr int HD = H * DK;  // 2048

  const float* x  = (const float*)d_in[0];
  const float* wq = (const float*)d_in[1];
  const float* wk = (const float*)d_in[2];
  const float* wv = (const float*)d_in[3];
  const float* wo = (const float*)d_in[4];
  const float* pe = (const float*)d_in[5];
  // d_in[6] = mask (zeros; reference applies causal itself) -- unused
  float* out = (float*)d_out;

  char* ws = (char*)d_ws;
  bf16* xb  = (bf16*)(ws + (size_t)0);
  bf16* wqb = (bf16*)(ws + ((size_t)16 << 20));
  bf16* wkb = (bf16*)(ws + ((size_t)24 << 20));
  bf16* wvb = (bf16*)(ws + ((size_t)32 << 20));
  bf16* wob = (bf16*)(ws + ((size_t)40 << 20));
  bf16* Qb  = (bf16*)(ws + ((size_t)48 << 20));
  bf16* Kb  = (bf16*)(ws + ((size_t)64 << 20));
  bf16* Vb  = (bf16*)(ws + ((size_t)80 << 20));
  bf16* Ab  = (bf16*)(ws + ((size_t)96 << 20));

  // 1) convert inputs to bf16
  {
    int n = M * E;
    f32_to_bf16_kernel<<<n / (4 * 256), 256, 0, stream>>>(x, xb, n);
    int nw = E * E;
    f32_to_bf16_kernel<<<nw / (4 * 256), 256, 0, stream>>>(wq, wqb, nw);
    f32_to_bf16_kernel<<<nw / (4 * 256), 256, 0, stream>>>(wk, wkb, nw);
    f32_to_bf16_kernel<<<nw / (4 * 256), 256, 0, stream>>>(wv, wvb, nw);
    f32_to_bf16_kernel<<<nw / (4 * 256), 256, 0, stream>>>(wo, wob, nw);
  }

  // 2) QKV projections: [M,E] x [HD,E]^T -> [M,HD] (bf16)
  dim3 ggrid(HD / 128, M / 128);
  gemm_nt_bf16<bf16><<<ggrid, 256, 0, stream>>>(xb, wqb, Qb, M, HD, E);
  gemm_nt_bf16<bf16><<<ggrid, 256, 0, stream>>>(xb, wkb, Kb, M, HD, E);
  gemm_nt_bf16<bf16><<<ggrid, 256, 0, stream>>>(xb, wvb, Vb, M, HD, E);

  // 3) RoPE on Q and K (in place, bf16)
  {
    int pairs = M * (HD / 2);
    rope_bf16_kernel<<<pairs / 256, 256, 0, stream>>>(Qb, pe, pairs);
    rope_bf16_kernel<<<pairs / 256, 256, 0, stream>>>(Kb, pe, pairs);
  }

  // 4) causal flash attention -> Ab (bf16 [M, HD])
  {
    size_t smem = (size_t)4 * 128 * 136 * sizeof(bf16);  // 136 KB dynamic LDS
    (void)hipFuncSetAttribute((const void*)flash_attn_kernel,
                              hipFuncAttributeMaxDynamicSharedMemorySize, (int)smem);
    dim3 fgrid(S / 128, B * H);
    flash_attn_kernel<<<fgrid, 256, smem, stream>>>(Qb, Kb, Vb, Ab, S);
  }

  // 5) output projection: [M,HD] x [E,HD]^T -> [M,E] (f32 -> d_out)
  dim3 ogrid(E / 128, M / 128);
  gemm_nt_bf16<float><<<ogrid, 256, 0, stream>>>(Ab, wob, out, M, E, HD);
}
